// SelfAttention_67027259621590
// MI455X (gfx1250) — matrix operand order repaired and checked by reading, not compiled, not run on hardware
//
#include <hip/hip_runtime.h>
#include <hip/hip_bf16.h>
#include <stdint.h>

typedef __attribute__((ext_vector_type(16))) __bf16 bf16x16;
typedef __attribute__((ext_vector_type(8)))  float  fx8;

struct alignas(16) U128 { unsigned int a, b, c, d; };
union ABFrag { bf16x16 v; U128 u[2]; };
union Pack8  { __bf16 h[8]; U128 u; };
union Pack4  { __bf16 h[4]; unsigned long long q; };

#define WMMA_BF16(A_, B_, C_)                                                  \
    __builtin_amdgcn_wmma_f32_16x16x32_bf16(false, (A_), false, (B_),          \
                                            (short)0, (C_), false, false)

// Async global -> LDS copy, 16B per lane. ASYNCcnt-tracked.
__device__ __forceinline__ void async_ld_b128(void* lds, const void* gptr) {
    unsigned loff = (unsigned)(uintptr_t)lds;  // low 32 bits = LDS offset
    asm volatile("global_load_async_to_lds_b128 %0, %1, off"
                 :: "v"(loff), "v"(gptr) : "memory");
}
#define WAIT_ASYNC(N) asm volatile("s_wait_asynccnt " #N ::: "memory")

// ---------------------------------------------------------------------------
// Elementwise fp32 -> bf16 (8 elements / thread)
// ---------------------------------------------------------------------------
__global__ __launch_bounds__(256) void cvt_f32_bf16(const float* __restrict__ in,
                                                    __bf16* __restrict__ out,
                                                    int n8) {
    const int i = blockIdx.x * 256 + threadIdx.x;
    if (i >= n8) return;
    const float4 f0 = *(const float4*)(in + (size_t)i * 8);
    const float4 f1 = *(const float4*)(in + (size_t)i * 8 + 4);
    Pack8 p;
    p.h[0] = (__bf16)f0.x; p.h[1] = (__bf16)f0.y;
    p.h[2] = (__bf16)f0.z; p.h[3] = (__bf16)f0.w;
    p.h[4] = (__bf16)f1.x; p.h[5] = (__bf16)f1.y;
    p.h[6] = (__bf16)f1.z; p.h[7] = (__bf16)f1.w;
    *(U128*)(out + (size_t)i * 8) = p.u;
}

// ---------------------------------------------------------------------------
// Transpose + convert: W[K][N] fp32 -> Wt[N][K] bf16. 32x32 tile / 256 thr.
// ---------------------------------------------------------------------------
__global__ __launch_bounds__(256) void transpose_cvt(const float* __restrict__ W,
                                                     __bf16* __restrict__ Wt,
                                                     int K, int N) {
    __shared__ float tile[32][33];
    const int t  = threadIdx.x;
    const int k0 = blockIdx.y * 32, n0 = blockIdx.x * 32;
    const int r  = t >> 3, c = (t & 7) * 4;
    const float4 f = *(const float4*)(W + (size_t)(k0 + r) * N + n0 + c);
    tile[r][c + 0] = f.x; tile[r][c + 1] = f.y;
    tile[r][c + 2] = f.z; tile[r][c + 3] = f.w;
    __syncthreads();
    Pack4 p;
    p.h[0] = (__bf16)tile[c + 0][r];
    p.h[1] = (__bf16)tile[c + 1][r];
    p.h[2] = (__bf16)tile[c + 2][r];
    p.h[3] = (__bf16)tile[c + 3][r];
    *(unsigned long long*)(Wt + (size_t)(n0 + r) * K + k0 + c) = p.q;
}

// ---------------------------------------------------------------------------
// GEMM: C[M,N] = A[M,K] @ Bt[N,K]^T, bf16 in, bf16 WMMA compute.
// Block 256 thr = 8 waves; block tile 256x128, BK=32; wave tile 64x64
// (4x4 WMMA tiles -> 16 WMMA per 16 ds_load_b128). Double-buffered LDS fed
// by async global->LDS copies (6 async ops per wave per tile).
// OUTMODE: 0 = fp32 row-major, 1 = bf16 row-major, 2 = bf16 transposed [N][M]
// ---------------------------------------------------------------------------
template <int OUTMODE>
__global__ __launch_bounds__(256) void gemm_bf16nt(const __bf16* __restrict__ A,
                                                   const __bf16* __restrict__ Bt,
                                                   void* __restrict__ Cv,
                                                   int M, int N, int K) {
    __shared__ __bf16 a_lds[2][256][40];
    __shared__ __bf16 b_lds[2][128][40];

    const int t    = threadIdx.x;
    const int w    = t >> 5;
    const int lane = t & 31;
    const int half = lane >> 4;
    const int l16  = lane & 15;
    const int wm   = w & 3;   // 4 row groups of 64
    const int wn   = w >> 2;  // 2 col groups of 64
    const int m0   = blockIdx.y * 256;
    const int n0   = blockIdx.x * 128;

    const fx8 zero = {0.f, 0.f, 0.f, 0.f, 0.f, 0.f, 0.f, 0.f};
    fx8 acc[4][4];
#pragma unroll
    for (int tm = 0; tm < 4; ++tm)
#pragma unroll
        for (int tn = 0; tn < 4; ++tn) acc[tm][tn] = zero;

    const int trow = t >> 2;        // 0..63
    const int tcol = (t & 3) * 8;   // 0,8,16,24

    auto issue_tile = [&](int k0, int buf) {
#pragma unroll
        for (int j = 0; j < 4; ++j)  // A: 256 rows
            async_ld_b128(&a_lds[buf][trow + j * 64][tcol],
                          A + (size_t)(m0 + trow + j * 64) * K + k0 + tcol);
#pragma unroll
        for (int j = 0; j < 2; ++j)  // Bt: 128 rows
            async_ld_b128(&b_lds[buf][trow + j * 64][tcol],
                          Bt + (size_t)(n0 + trow + j * 64) * K + k0 + tcol);
    };

    issue_tile(0, 0);
    int cur = 0;
    const int niter = K >> 5;
    for (int i = 0; i < niter; ++i) {
        if (i + 1 < niter) {
            issue_tile((i + 1) << 5, cur ^ 1);
            WAIT_ASYNC(6);          // tile i's 6 per-wave async ops complete
        } else {
            WAIT_ASYNC(0);
        }
        __syncthreads();

        ABFrag bfr[4];
#pragma unroll
        for (int tn = 0; tn < 4; ++tn) {
            const __bf16* rp = &b_lds[cur][wn * 64 + tn * 16 + l16][0];
            bfr[tn].u[0] = *(const U128*)(rp + half * 16);
            bfr[tn].u[1] = *(const U128*)(rp + half * 16 + 8);
        }
#pragma unroll
        for (int tm = 0; tm < 4; ++tm) {
            ABFrag af;
            const __bf16* rp = &a_lds[cur][wm * 64 + tm * 16 + l16][0];
            af.u[0] = *(const U128*)(rp + half * 16);
            af.u[1] = *(const U128*)(rp + half * 16 + 8);
#pragma unroll
            for (int tn = 0; tn < 4; ++tn)
                acc[tm][tn] = WMMA_BF16(af.v, bfr[tn].v, acc[tm][tn]);
        }

        __syncthreads();
        cur ^= 1;
    }

#pragma unroll
    for (int tm = 0; tm < 4; ++tm) {
        const int rbase = m0 + wm * 64 + tm * 16 + half * 8;
#pragma unroll
        for (int tn = 0; tn < 4; ++tn) {
            const int col = n0 + wn * 64 + tn * 16 + l16;
            if (OUTMODE == 0) {
                float* C = (float*)Cv;
#pragma unroll
                for (int r = 0; r < 8; ++r)
                    C[(size_t)(rbase + r) * N + col] = acc[tm][tn][r];
            } else if (OUTMODE == 1) {
                __bf16* C = (__bf16*)Cv;
#pragma unroll
                for (int r = 0; r < 8; ++r)
                    C[(size_t)(rbase + r) * N + col] = (__bf16)acc[tm][tn][r];
            } else {
                __bf16* C = (__bf16*)Cv;  // [N][M], one b128 per tile per lane
                Pack8 p;
#pragma unroll
                for (int r = 0; r < 8; ++r) p.h[r] = (__bf16)acc[tm][tn][r];
                *(U128*)(C + (size_t)col * M + rbase) = p.u;
            }
        }
    }
}

// ---------------------------------------------------------------------------
// Flash attention (causal, GQA), all-bf16, double-buffered async K/V blocks.
// Q: [B*S,4096]; K: [B*S,1024]; Vt: [1024][B*S] (transposed); O: [B*S,4096].
// Block 128 thr = 4 waves; (b, h, 64 q rows) per block; 32-key inner blocks.
// ---------------------------------------------------------------------------
__global__ __launch_bounds__(128) void attn_flash(const __bf16* __restrict__ Q,
                                                  const __bf16* __restrict__ Kb,
                                                  const __bf16* __restrict__ Vt,
                                                  __bf16* __restrict__ O,
                                                  int S) {
    __shared__ __bf16 k_lds[2][32][136];    // [kv][d]
    __shared__ __bf16 vt_lds[2][128][40];   // [d][kv]
    __shared__ __bf16 p_lds[4][16][40];     // per-wave P transpose scratch

    const int t    = threadIdx.x;
    const int w    = t >> 5;
    const int lane = t & 31;
    const int half = lane >> 4;
    const int l16  = lane & 15;
    const int qb   = blockIdx.x;
    const int h    = blockIdx.y;
    const int b    = blockIdx.z;
    const int kvh  = h >> 2;
    const float scale = 0.08838834764831845f;  // 1/sqrt(128)

    // Q A-fragments straight from global bf16 (2 x b128 per chunk per lane).
    ABFrag qf[4];
    {
        const int qrow = b * S + qb * 64 + w * 16 + l16;
        const __bf16* qrp = Q + (size_t)qrow * 4096 + h * 128;
#pragma unroll
        for (int dc = 0; dc < 4; ++dc) {
            qf[dc].u[0] = *(const U128*)(qrp + dc * 32 + half * 16);
            qf[dc].u[1] = *(const U128*)(qrp + dc * 32 + half * 16 + 8);
        }
    }

    auto issue_kv = [&](int kb, int buf) {
        const int cc = (t & 15) * 8;
#pragma unroll
        for (int i2 = 0; i2 < 4; ++i2) {
            const int row = (t >> 4) + i2 * 8;
            async_ld_b128(&k_lds[buf][row][cc],
                          Kb + (size_t)(b * S + kb * 32 + row) * 1024 +
                              kvh * 128 + cc);
        }
#pragma unroll
        for (int j = 0; j < 4; ++j) {
            async_ld_b128(&vt_lds[buf][t][j * 8],
                          Vt + (size_t)(kvh * 128 + t) * 4096 + b * S +
                              kb * 32 + j * 8);
        }
    };

    const fx8 zero = {0.f, 0.f, 0.f, 0.f, 0.f, 0.f, 0.f, 0.f};
    fx8 o[8];
#pragma unroll
    for (int nt = 0; nt < 8; ++nt) o[nt] = zero;
    float m[8], l[8];
#pragma unroll
    for (int r = 0; r < 8; ++r) { m[r] = -3.0e38f; l[r] = 0.f; }

    const int kb_end = 2 * qb + 2;
    issue_kv(0, 0);
    int cur = 0;
    for (int kb = 0; kb < kb_end; ++kb) {
        if (kb + 1 < kb_end) {
            issue_kv(kb + 1, cur ^ 1);
            WAIT_ASYNC(8);          // block kb's 8 per-wave async ops complete
        } else {
            WAIT_ASYNC(0);
        }
        __syncthreads();

        // S = Q K^T : two 16x16 tiles over 32 kv cols, K-dim 128 (4 WMMAs ea).
        fx8 s0 = zero, s1 = zero;
#pragma unroll
        for (int dc = 0; dc < 4; ++dc) {
            ABFrag k0f, k1f;
            const __bf16* rp0 = &k_lds[cur][l16][dc * 32];
            k0f.u[0] = *(const U128*)(rp0 + half * 16);
            k0f.u[1] = *(const U128*)(rp0 + half * 16 + 8);
            const __bf16* rp1 = &k_lds[cur][16 + l16][dc * 32];
            k1f.u[0] = *(const U128*)(rp1 + half * 16);
            k1f.u[1] = *(const U128*)(rp1 + half * 16 + 8);
            s0 = WMMA_BF16(qf[dc].v, k0f.v, s0);
            s1 = WMMA_BF16(qf[dc].v, k1f.v, s1);
        }

        // Online softmax per row (lane holds rows half*8 + r at col l16).
#pragma unroll
        for (int r = 0; r < 8; ++r) {
            const int qrow = qb * 64 + w * 16 + half * 8 + r;
            const int kv0  = kb * 32 + l16;
            float v0 = s0[r] * scale;
            float v1 = s1[r] * scale;
            if (kv0 > qrow)      v0 = -3.0e38f;
            if (kv0 + 16 > qrow) v1 = -3.0e38f;
            float mx = fmaxf(v0, v1);
            mx = fmaxf(mx, __shfl_xor(mx, 1, 32));
            mx = fmaxf(mx, __shfl_xor(mx, 2, 32));
            mx = fmaxf(mx, __shfl_xor(mx, 4, 32));
            mx = fmaxf(mx, __shfl_xor(mx, 8, 32));
            const float mn    = fmaxf(m[r], mx);
            const float alpha = __expf(m[r] - mn);
            const float p0 = __expf(v0 - mn);
            const float p1 = __expf(v1 - mn);
            float rs = p0 + p1;
            rs += __shfl_xor(rs, 1, 32);
            rs += __shfl_xor(rs, 2, 32);
            rs += __shfl_xor(rs, 4, 32);
            rs += __shfl_xor(rs, 8, 32);
            m[r] = mn;
            l[r] = l[r] * alpha + rs;
#pragma unroll
            for (int nt = 0; nt < 8; ++nt) o[nt][r] *= alpha;
            p_lds[w][half * 8 + r][l16]      = (__bf16)p0;
            p_lds[w][half * 8 + r][16 + l16] = (__bf16)p1;
        }

        // O += P @ V (K-dim = 32 kv rows).
        ABFrag pf;
        const __bf16* prp = &p_lds[w][l16][0];
        pf.u[0] = *(const U128*)(prp + half * 16);
        pf.u[1] = *(const U128*)(prp + half * 16 + 8);
#pragma unroll
        for (int nt = 0; nt < 8; ++nt) {
            ABFrag vf;
            const __bf16* vrp = &vt_lds[cur][nt * 16 + l16][0];
            vf.u[0] = *(const U128*)(vrp + half * 16);
            vf.u[1] = *(const U128*)(vrp + half * 16 + 8);
            o[nt] = WMMA_BF16(pf.v, vf.v, o[nt]);
        }

        __syncthreads();  // all reads of buf `cur` done before it is refilled
        cur ^= 1;
    }

    // Normalize and store bf16.
#pragma unroll
    for (int r = 0; r < 8; ++r) {
        const float inv  = 1.0f / l[r];
        const int   qrow = b * S + qb * 64 + w * 16 + half * 8 + r;
        __bf16* orow = O + (size_t)qrow * 4096 + h * 128;
#pragma unroll
        for (int nt = 0; nt < 8; ++nt)
            orow[nt * 16 + l16] = (__bf16)(o[nt][r] * inv);
    }
}

// ---------------------------------------------------------------------------
extern "C" void kernel_launch(void* const* d_in, const int* in_sizes, int n_in,
                              void* d_out, int out_size, void* d_ws,
                              size_t ws_size, hipStream_t stream) {
    const float* x  = (const float*)d_in[0];
    const float* wq = (const float*)d_in[1];
    const float* wk = (const float*)d_in[2];
    const float* wv = (const float*)d_in[3];
    const float* wo = (const float*)d_in[4];
    float* out = (float*)d_out;

    const int B = 2, S = 2048, DIM = 4096, KVD = 1024;
    const int M = B * S;  // 4096
    const size_t SZ_BIG = (size_t)M * DIM;  // 16M
    const size_t SZ_KV  = (size_t)M * KVD;  // 4M

    __bf16* xb   = (__bf16*)d_ws;
    __bf16* wqt  = xb + SZ_BIG;
    __bf16* wkt  = wqt + SZ_BIG;
    __bf16* wvt  = wkt + SZ_KV;
    __bf16* wot  = wvt + SZ_KV;
    __bf16* q    = wot + SZ_BIG;
    __bf16* kbuf = q + SZ_BIG;
    __bf16* vT   = kbuf + SZ_KV;
    __bf16* attn = vT + SZ_KV;

    // One-time precision/layout conversion (bf16 everywhere, weights ^T).
    cvt_f32_bf16<<<dim3((int)(SZ_BIG / 8 / 256)), dim3(256), 0, stream>>>(
        x, xb, (int)(SZ_BIG / 8));
    transpose_cvt<<<dim3(DIM / 32, DIM / 32), dim3(256), 0, stream>>>(
        wq, wqt, DIM, DIM);
    transpose_cvt<<<dim3(KVD / 32, DIM / 32), dim3(256), 0, stream>>>(
        wk, wkt, DIM, KVD);
    transpose_cvt<<<dim3(KVD / 32, DIM / 32), dim3(256), 0, stream>>>(
        wv, wvt, DIM, KVD);
    transpose_cvt<<<dim3(DIM / 32, DIM / 32), dim3(256), 0, stream>>>(
        wo, wot, DIM, DIM);

    // Projections (bf16 out; V stored transposed for the attention B-frags).
    gemm_bf16nt<1><<<dim3(DIM / 128, M / 256), dim3(256), 0, stream>>>(
        xb, wqt, q, M, DIM, DIM);
    gemm_bf16nt<1><<<dim3(KVD / 128, M / 256), dim3(256), 0, stream>>>(
        xb, wkt, kbuf, M, KVD, DIM);
    gemm_bf16nt<2><<<dim3(KVD / 128, M / 256), dim3(256), 0, stream>>>(
        xb, wvt, vT, M, KVD, DIM);

    attn_flash<<<dim3(S / 64, 32, B), dim3(128), 0, stream>>>(
        q, kbuf, vT, attn, S);

    // Output projection -> fp32 d_out.
    gemm_bf16nt<0><<<dim3(DIM / 128, M / 256), dim3(256), 0, stream>>>(
        attn, wot, out, M, DIM, DIM);
}
